// BiLSTM_CRF_7129645711621
// MI455X (gfx1250) — matrix operand order repaired and checked by reading, not compiled, not run on hardware
//
#include <hip/hip_runtime.h>
#include <math.h>

typedef __attribute__((ext_vector_type(16))) _Float16 v16h;
typedef __attribute__((ext_vector_type(8)))  _Float16 v8h;
typedef __attribute__((ext_vector_type(8)))  float    v8f;

#define BATCH 1024
#define LSEQ  512
#define NC    48
#define CSTART 46
#define CSTOP  47
#define KPAD  64

#define EXP2(x) __builtin_amdgcn_exp2f(x)
#define LOG2(x) __builtin_amdgcn_logf(x)
#define INV_LN2 1.4426950408889634f
#define LN2     0.6931471805599453f

// ---------------------------------------------------------------------------
// Forward algorithm (log-partition) via rescaled linear-space WMMA recursion,
// carried in log2 domain so exp/log are raw v_exp_f32 / v_log_f32.
// Block = 96 threads = 3 waves; block owns batch rows [16*blockIdx, +16);
// wave w owns state columns [16w, 16w+16). alpha kept in D-fragment regs.
// Emission loads are software-pipelined one step ahead (single clause, no
// divergent per-load waits); the mask blend is a register select.
// ---------------------------------------------------------------------------
__launch_bounds__(96)
__global__ void crf_partition_kernel(const float* __restrict__ em,
                                     const float* __restrict__ T,
                                     const float* __restrict__ mask,
                                     float* __restrict__ part)
{
    __shared__ _Float16 aexp[16][KPAD];   // exp2(alpha2 - m2), K-padded to 64
    __shared__ float    wmax[16][4];      // per-wave row maxes
    __shared__ float    wsum[16][4];      // per-wave partial exp sums (final)
    __shared__ float    lenf[16];         // per-row sequence lengths

    const int tid  = threadIdx.x;
    const int w    = tid >> 5;        // wave id 0..2  (N tile)
    const int l    = tid & 31;        // lane in wave
    const int ln   = l & 15;
    const int hi   = l >> 4;
    const int hi8  = hi * 8;
    const int bbase = blockIdx.x * 16;
    const int n    = w * 16 + ln;     // this lane's state column

    // zero the K padding halfs [48,64) of every aexp row (written once)
    for (int i = tid; i < 128; i += 96) {
        int row = i >> 3, dk = i & 7;
        *(unsigned int*)&aexp[row][48 + dk * 2] = 0u;
    }
    if (tid < 16) lenf[tid] = 0.0f;
    __syncthreads();

    // cooperative per-row length = sum(mask[row]) (exact: 0/1 floats)
    {
        int r  = tid % 16;
        int c0 = tid / 16;            // 0..5
        const float* mrow = mask + (size_t)(bbase + r) * LSEQ;
        float s = 0.0f;
        for (int t = c0; t < LSEQ; t += 6) s += mrow[t];
        atomicAdd(&lenf[r], s);
    }

    // Constant B fragments: exp2(T/ln2) == exp(T) in f16, B-layout:
    // lane holds column n; half h -> K = kc*32 + (l/16)*16 + h. K>=48 pad 0.
    v16h bfrag0, bfrag1;
#pragma unroll
    for (int h = 0; h < 16; ++h) {
        int k0 = hi * 16 + h;
        int k1 = 32 + hi * 16 + h;
        float t0 = (k0 < NC) ? T[k0 * NC + n] : -10000.0f;
        float t1 = (k1 < NC) ? T[k1 * NC + n] : -10000.0f;
        bfrag0[h] = (_Float16)EXP2(t0 * INV_LN2);
        bfrag1[h] = (_Float16)EXP2(t1 * INV_LN2);
    }

    const float tstart2 = T[CSTART * NC + n] * INV_LN2;
    const float tstop2  = T[n * NC + CSTOP] * INV_LN2;

    // alpha (log2 domain) in D-layout: alpha[v] = row r = v + 8*hi, column n
    float alpha[8];
#pragma unroll
    for (int v = 0; v < 8; ++v) {
        size_t b = (size_t)(bbase + hi8 + v);
        alpha[v] = em[(b * LSEQ + 0) * NC + n] * INV_LN2 + tstart2;
    }
    __syncthreads();

    int lenv[8];
    int maxL = 0;
#pragma unroll
    for (int v = 0; v < 8; ++v) lenv[v] = (int)(lenf[hi8 + v] + 0.5f);
    for (int r = 0; r < 16; ++r) {
        int lr = (int)(lenf[r] + 0.5f);
        maxL = (lr > maxL) ? lr : maxL;
    }

    // pipelined emission loads: one clause of 8 per step
    float ecur[8] = {0.f, 0.f, 0.f, 0.f, 0.f, 0.f, 0.f, 0.f};
    if (maxL > 1) {
#pragma unroll
        for (int v = 0; v < 8; ++v) {
            size_t b = (size_t)(bbase + hi8 + v);
            ecur[v] = em[(b * LSEQ + 1) * NC + n];
        }
    }

    for (int t = 1; t < maxL; ++t) {
        // --- prefetch emissions for t+1 (uniform branch, full-iter cover) ---
        float enext[8] = {0.f, 0.f, 0.f, 0.f, 0.f, 0.f, 0.f, 0.f};
        if (t + 1 < maxL) {
#pragma unroll
            for (int v = 0; v < 8; ++v) {
                size_t b = (size_t)(bbase + hi8 + v);
                enext[v] = em[(b * LSEQ + t + 1) * NC + n];
            }
        }

        // --- wave-local row max (reduce over 16-lane group) ---
        float lm[8];
#pragma unroll
        for (int v = 0; v < 8; ++v) {
            float m = alpha[v];
            m = fmaxf(m, __shfl_xor(m, 1, 32));
            m = fmaxf(m, __shfl_xor(m, 2, 32));
            m = fmaxf(m, __shfl_xor(m, 4, 32));
            m = fmaxf(m, __shfl_xor(m, 8, 32));
            lm[v] = m;
        }
        if (ln == 0) {
#pragma unroll
            for (int v = 0; v < 8; ++v) wmax[hi8 + v][w] = lm[v];
        }
        __syncthreads();

        // --- combine maxes across 3 waves, write exp2(alpha-m) to LDS ---
        float m[8];
#pragma unroll
        for (int v = 0; v < 8; ++v) {
            int r = hi8 + v;
            m[v] = fmaxf(wmax[r][0], fmaxf(wmax[r][1], wmax[r][2]));
            aexp[r][n] = (_Float16)EXP2(alpha[v] - m[v]);
        }
        __syncthreads();

        // --- rebuild full-K A fragments (row = ln), 4x ds_load_b128 ---
        const _Float16* arow = &aexp[ln][0];
        v8h p0 = *(const v8h*)(arow + 2 * hi8);
        v8h p1 = *(const v8h*)(arow + 8 + 2 * hi8);
        v8h p2 = *(const v8h*)(arow + 32 + 2 * hi8);
        v8h p3 = *(const v8h*)(arow + 40 + 2 * hi8);
        v16h a0, a1;
#pragma unroll
        for (int i = 0; i < 8; ++i) {
            a0[i] = p0[i]; a0[i + 8] = p1[i];
            a1[i] = p2[i]; a1[i + 8] = p3[i];
        }

        v8f acc = {};
        acc = __builtin_amdgcn_wmma_f32_16x16x32_f16(false, a0, false, bfrag0,
                                                     (short)0, acc, false, false);
        acc = __builtin_amdgcn_wmma_f32_16x16x32_f16(false, a1, false, bfrag1,
                                                     (short)0, acc, false, false);

        // --- alpha update; rows past their length frozen via select ---
#pragma unroll
        for (int v = 0; v < 8; ++v) {
            float nv = m[v] + LOG2(acc[v]) + ecur[v] * INV_LN2;
            alpha[v] = (t < lenv[v]) ? nv : alpha[v];
        }
#pragma unroll
        for (int v = 0; v < 8; ++v) ecur[v] = enext[v];
    }

    // --- final: part = ln2 * log2sumexp2_j(alpha + tstop2) ---
    float fv[8], lm2[8];
#pragma unroll
    for (int v = 0; v < 8; ++v) {
        float x = alpha[v] + tstop2;
        fv[v] = x;
        x = fmaxf(x, __shfl_xor(x, 1, 32));
        x = fmaxf(x, __shfl_xor(x, 2, 32));
        x = fmaxf(x, __shfl_xor(x, 4, 32));
        x = fmaxf(x, __shfl_xor(x, 8, 32));
        lm2[v] = x;
    }
    if (ln == 0) {
#pragma unroll
        for (int v = 0; v < 8; ++v) wmax[hi8 + v][w] = lm2[v];
    }
    __syncthreads();
#pragma unroll
    for (int v = 0; v < 8; ++v) {
        int r = hi8 + v;
        float mm = fmaxf(wmax[r][0], fmaxf(wmax[r][1], wmax[r][2]));
        float s = EXP2(fv[v] - mm);
        s += __shfl_xor(s, 1, 32);
        s += __shfl_xor(s, 2, 32);
        s += __shfl_xor(s, 4, 32);
        s += __shfl_xor(s, 8, 32);
        if (ln == 0) wsum[r][w] = s;
    }
    __syncthreads();
    if (w == 0 && l < 16) {
        int r = l;
        float mm = fmaxf(wmax[r][0], fmaxf(wmax[r][1], wmax[r][2]));
        float ss = wsum[r][0] + wsum[r][1] + wsum[r][2];
        part[bbase + r] = LN2 * (mm + LOG2(ss));
    }
}

// ---------------------------------------------------------------------------
// Gold path score: one wave per batch row, lane-strided over time.
// ---------------------------------------------------------------------------
__launch_bounds__(256)
__global__ void crf_gold_kernel(const float* __restrict__ em,
                                const float* __restrict__ T,
                                const int* __restrict__ tags,
                                const float* __restrict__ mask,
                                float* __restrict__ gold)
{
    const int b    = (blockIdx.x * blockDim.x + threadIdx.x) >> 5;
    const int lane = threadIdx.x & 31;
    if (b >= BATCH) return;

    const int*   tg   = tags + (size_t)b * LSEQ;
    const float* mrow = mask + (size_t)b * LSEQ;
    const float* erow = em + (size_t)b * LSEQ * NC;

    float s = 0.0f, msum = 0.0f;
    for (int t = lane; t < LSEQ; t += 32) {
        int   tt = tg[t];
        float mk = mrow[t];
        msum += mk;
        if (t == 0) {
            s += erow[tt] + T[CSTART * NC + tt];
        } else {
            int tp = tg[t - 1];
            s += mk * (erow[t * NC + tt] + T[tp * NC + tt]);
        }
    }
#pragma unroll
    for (int off = 16; off > 0; off >>= 1) {
        s    += __shfl_xor(s, off, 32);
        msum += __shfl_xor(msum, off, 32);
    }
    if (lane == 0) {
        int last = tg[(int)(msum + 0.5f) - 1];   // length >= 1 guaranteed
        gold[b] = s + T[last * NC + CSTOP];
    }
}

// ---------------------------------------------------------------------------
// mean(partition - gold)
// ---------------------------------------------------------------------------
__global__ void crf_reduce_kernel(const float* __restrict__ part,
                                  const float* __restrict__ gold,
                                  float* __restrict__ out)
{
    __shared__ float sbuf[256];
    int tid = threadIdx.x;
    float s = 0.0f;
    for (int i = tid; i < BATCH; i += 256) s += part[i] - gold[i];
    sbuf[tid] = s;
    __syncthreads();
    for (int off = 128; off > 0; off >>= 1) {
        if (tid < off) sbuf[tid] += sbuf[tid + off];
        __syncthreads();
    }
    if (tid == 0) out[0] = sbuf[0] * (1.0f / (float)BATCH);
}

extern "C" void kernel_launch(void* const* d_in, const int* in_sizes, int n_in,
                              void* d_out, int out_size, void* d_ws, size_t ws_size,
                              hipStream_t stream)
{
    const float* em   = (const float*)d_in[0];  // (B, L, C) f32
    const float* T    = (const float*)d_in[1];  // (C, C)    f32
    const int*   tags = (const int*)d_in[2];    // (B, L)    i32
    const float* mask = (const float*)d_in[3];  // (B, L)    f32

    float* part = (float*)d_ws;
    float* gold = part + BATCH;

    crf_partition_kernel<<<BATCH / 16, 96, 0, stream>>>(em, T, mask, part);
    crf_gold_kernel<<<BATCH / 8, 256, 0, stream>>>(em, T, tags, mask, gold);
    crf_reduce_kernel<<<1, 256, 0, stream>>>(part, gold, (float*)d_out);
}
